// FuzzyLayerOriginal_31542239822467
// MI455X (gfx1250) — matrix operand order repaired and checked by reading, not compiled, not run on hardware
//
#include <hip/hip_runtime.h>
#include <hip/hip_bf16.h>
#include <math.h>

typedef __attribute__((ext_vector_type(16))) _Float16 v16h;
typedef __attribute__((ext_vector_type(8)))  float    v8f;

#define NPIX (16 * 256 * 256)   // 1,048,576 pixels per channel
#define INV_NPIX (1.0f / 1048576.0f)
#define EPS 1e-5f

// ---------------------------------------------------------------------------
// Kernel 0: zero the stat accumulators (ws is poisoned 0xAA, not re-zeroed)
// ---------------------------------------------------------------------------
__global__ void k_zero(float* __restrict__ p, int n) {
    int i = blockIdx.x * blockDim.x + threadIdx.x;
    if (i < n) p[i] = 0.0f;
}

// ---------------------------------------------------------------------------
// Kernel 1: conv1 (64->1 ch, 3x3 SAME) + fuzzy exp(-sum d^2) -> z_raw,
//           plus global sum / sumsq for bn1.
// TY=4 vertical coarsening: each thread computes 4 pixels in a column strip.
// Per channel: 6 rows, each row's 3 contiguous taps load as one b96 ->
// 6 VMEM instr / channel / 4 pixels (vs 36), and 4 independent FMA chains.
// ---------------------------------------------------------------------------
__global__ void k_conv1_fuzzy(const float* __restrict__ x,
                              const float* __restrict__ w1,   // (1,64,3,3)
                              const float* __restrict__ b1,
                              const float* __restrict__ mu,   // row 0: 4 vals
                              const float* __restrict__ sg,   // row 0: 4 vals
                              float* __restrict__ zr,
                              float* __restrict__ stat) {     // [sum, sumsq]
    __shared__ float w1s[576];
    for (int i = threadIdx.x; i < 576; i += blockDim.x) w1s[i] = w1[i];
    __syncthreads();

    const int tid = blockIdx.x * blockDim.x + threadIdx.x;  // 0..262143
    const int xx  = tid & 255;          // lane-consecutive -> coalesced
    const int yg  = (tid >> 8) & 63;
    const int n   = tid >> 14;
    const int y0  = yg << 2;            // 4 rows: y0..y0+3

    const float m0 = mu[0], m1 = mu[1], m2 = mu[2], m3 = mu[3];
    const float r0 = 1.0f / sg[0], r1 = 1.0f / sg[1];
    const float r2 = 1.0f / sg[2], r3 = 1.0f / sg[3];
    const float bias = b1[0];

    float acc[4];
#pragma unroll
    for (int t = 0; t < 4; ++t) acc[t] = bias;

    const bool xin = (xx >= 1) & (xx <= 254);

    for (int c = 0; c < 64; ++c) {
        const float* xp = x + (((size_t)(n * 64 + c)) << 16) + xx;
        const float* wr = w1s + c * 9;
#pragma unroll
        for (int r = 0; r < 6; ++r) {
            const int yy = y0 + r - 1;
            // rows 1..4 are always in range (y0 in [0,252])
            const bool okY = (r >= 1 && r <= 4) || ((unsigned)yy < 256u);
            float lv = 0.0f, mv = 0.0f, rv = 0.0f;
            if (okY) {
                const float* row = xp + (yy << 8);
                if (xin) {              // contiguous -> global_load_b96
                    lv = row[-1]; mv = row[0]; rv = row[1];
                } else {
                    mv = row[0];
                    if (xx > 0)   lv = row[-1];
                    if (xx < 255) rv = row[1];
                }
            }
            // row r feeds pixel ty with tap ky = r - ty, ky in [0,3)
#pragma unroll
            for (int ty = 0; ty < 4; ++ty) {
                const int ky = r - ty;
                if (ky >= 0 && ky < 3) {
                    acc[ty] = fmaf(lv, wr[ky * 3 + 0], acc[ty]);
                    acc[ty] = fmaf(mv, wr[ky * 3 + 1], acc[ty]);
                    acc[ty] = fmaf(rv, wr[ky * 3 + 2], acc[ty]);
                }
            }
        }
    }

    float lsum = 0.0f, lsq = 0.0f;
#pragma unroll
    for (int ty = 0; ty < 4; ++ty) {
        const float h  = acc[ty];
        const float d0 = (h - m0) * r0;
        const float d1 = (h - m1) * r1;
        const float d2 = (h - m2) * r2;
        const float d3 = (h - m3) * r3;
        const float z  = __expf(-(d0 * d0 + d1 * d1 + d2 * d2 + d3 * d3));
        zr[(n << 16) + ((y0 + ty) << 8) + xx] = z;
        lsum += z;
        lsq = fmaf(z, z, lsq);
    }

    // wave32 reduction, then per-block reduction, then 2 atomics
#pragma unroll
    for (int off = 16; off; off >>= 1) {
        lsum += __shfl_xor(lsum, off, 32);
        lsq  += __shfl_xor(lsq,  off, 32);
    }
    __shared__ float ss[8], sq[8];
    const int lane = threadIdx.x & 31, wv = threadIdx.x >> 5;
    if (lane == 0) { ss[wv] = lsum; sq[wv] = lsq; }
    __syncthreads();
    if (threadIdx.x == 0) {
        float a = 0.0f, b = 0.0f;
#pragma unroll
        for (int i = 0; i < 8; ++i) { a += ss[i]; b += sq[i]; }
        atomicAdd(&stat[0], a);
        atomicAdd(&stat[1], b);
    }
}

// ---------------------------------------------------------------------------
// Kernel 2: bn1 finalize -> z = z_raw * s1 + t1
// ---------------------------------------------------------------------------
__global__ void k_bn1_fin(const float* __restrict__ stat,
                          const float* __restrict__ g,
                          const float* __restrict__ b,
                          float* __restrict__ s1t1) {
    if (threadIdx.x == 0) {
        const float mean = stat[0] * INV_NPIX;
        const float var  = stat[1] * INV_NPIX - mean * mean;
        const float s    = g[0] * rsqrtf(var + EPS);
        s1t1[0] = s;
        s1t1[1] = b[0] - mean * s;
    }
}

// ---------------------------------------------------------------------------
// Kernel 3: 9-tap shift-sum S_k and Gram matrix G_kl = sum_p z_k z_l over the
// bn1-affine z (channel-independent!). bn2 stats follow analytically:
//   mean_c  = (w_c . S) / NP          (conv2_b cancels under bn)
//   E[v^2]_c = (w_c^T G w_c) / NP
// 54 FMA/pixel instead of 64ch x 11 ops/pixel (~11x cheaper).
// gsum layout: [0..9) = S, [9..54) = upper-triangle G (k<=l).
// ---------------------------------------------------------------------------
__global__ void k_zgram(const float* __restrict__ zr,
                        const float* __restrict__ s1t1,
                        float* __restrict__ gsum) {
    __shared__ float red[54];
    for (int i = threadIdx.x; i < 54; i += blockDim.x) red[i] = 0.0f;
    __syncthreads();

    const float s1 = s1t1[0], t1 = s1t1[1];

    float S[9], G[45];
#pragma unroll
    for (int i = 0; i < 9; ++i) S[i] = 0.0f;
#pragma unroll
    for (int i = 0; i < 45; ++i) G[i] = 0.0f;

    for (int p = blockIdx.x * blockDim.x + threadIdx.x; p < NPIX;
         p += gridDim.x * blockDim.x) {
        const int n  = p >> 16;
        const int y  = (p >> 8) & 255;
        const int xx = p & 255;
        const float* zp = zr + ((size_t)n << 16);

        float t[9];
#pragma unroll
        for (int ky = 0; ky < 3; ++ky) {
            const int yy = y + ky - 1;
            const bool oky = (unsigned)yy < 256u;
#pragma unroll
            for (int kx = 0; kx < 3; ++kx) {
                const int xv = xx + kx - 1;
                float v = 0.0f;   // SAME zero padding (outside tensor)
                if (oky && (unsigned)xv < 256u)
                    v = fmaf(zp[(yy << 8) + xv], s1, t1);
                t[ky * 3 + kx] = v;
            }
        }
#pragma unroll
        for (int k = 0; k < 9; ++k) {
            S[k] += t[k];
#pragma unroll
            for (int l = k; l < 9; ++l) {
                const int idx = k * 9 - (k * (k - 1)) / 2 + (l - k);
                G[idx] = fmaf(t[k], t[l], G[idx]);
            }
        }
    }

    // wave32 shuffle reduction of all 54 accumulators
#pragma unroll
    for (int i = 0; i < 9; ++i)
#pragma unroll
        for (int off = 16; off; off >>= 1) S[i] += __shfl_xor(S[i], off, 32);
#pragma unroll
    for (int i = 0; i < 45; ++i)
#pragma unroll
        for (int off = 16; off; off >>= 1) G[i] += __shfl_xor(G[i], off, 32);

    const int lane = threadIdx.x & 31;
    if (lane == 0) {
#pragma unroll
        for (int i = 0; i < 9; ++i)  atomicAdd(&red[i],     S[i]);
#pragma unroll
        for (int i = 0; i < 45; ++i) atomicAdd(&red[9 + i], G[i]);
    }
    __syncthreads();
    if (threadIdx.x < 54) atomicAdd(&gsum[threadIdx.x], red[threadIdx.x]);
}

// ---------------------------------------------------------------------------
// Kernel 4: bn2 finalize from S/G -> out = conv2 * sc[c] + sh[c]
// ---------------------------------------------------------------------------
__global__ void k_bn2_fin(const float* __restrict__ gsum,
                          const float* __restrict__ w2,   // (64,1,3,3)
                          const float* __restrict__ g,
                          const float* __restrict__ b,
                          float* __restrict__ sc,
                          float* __restrict__ sh) {
    const int c = threadIdx.x;
    if (c >= 64) return;
    float w[9];
#pragma unroll
    for (int k = 0; k < 9; ++k) w[k] = w2[c * 9 + k];

    float sm = 0.0f;
#pragma unroll
    for (int k = 0; k < 9; ++k) sm = fmaf(w[k], gsum[k], sm);
    const float mean = sm * INV_NPIX;

    float e2 = 0.0f;
#pragma unroll
    for (int k = 0; k < 9; ++k)
#pragma unroll
        for (int l = 0; l < 9; ++l) {
            const int kk = (k < l) ? k : l;
            const int ll = (k < l) ? l : k;
            const int idx = 9 + kk * 9 - (kk * (kk - 1)) / 2 + (ll - kk);
            e2 = fmaf(w[k] * w[l], gsum[idx], e2);
        }
    const float var = e2 * INV_NPIX - mean * mean;
    const float s   = g[c] * rsqrtf(var + EPS);
    sc[c] = s;
    sh[c] = b[c] - mean * s;
}

// ---------------------------------------------------------------------------
// Kernel 5: conv2 + bn2 fused via V_WMMA_F32_16X16X32_F16.
// GEMM: D[ch, px] = sum_k A[ch, k] * B[k, px],  A = w2 (f16, K padded 9->32),
// B = im2col(z) patch (f16). One wave = 16 pixels x 64 channels (4 M-tiles,
// B reused). Pixels in N -> coalesced 64B store segments (write-bound kernel).
// ---------------------------------------------------------------------------
__global__ void k_conv2_wmma(const float* __restrict__ zr,
                             const float* __restrict__ w2,
                             const float* __restrict__ s1t1,
                             const float* __restrict__ sc,
                             const float* __restrict__ sh,
                             float* __restrict__ out) {
    const int lane = threadIdx.x & 31;
    const int wv   = threadIdx.x >> 5;
    const int tile = blockIdx.x * (blockDim.x >> 5) + wv;  // 65536 tiles total
    const int p0   = tile << 4;                            // 16 pixels / tile
    const int n    = p0 >> 16;
    const int y    = (p0 >> 8) & 255;
    const int x0   = p0 & 255;                             // multiple of 16

    const float s1 = s1t1[0], t1 = s1t1[1];

    // ---- B matrix (32 x 16, f16): lanes 0-15 hold column N=lane with
    // K=0..15 packed in halves 0..15; lanes 16-31 hold K=16..31 (all zero).
    v16h bM;
#pragma unroll
    for (int i = 0; i < 16; ++i) bM[i] = (_Float16)0.0f;
    if (lane < 16) {
        const int px = x0 + lane;
        const float* zp = zr + ((size_t)n << 16);
#pragma unroll
        for (int ky = 0; ky < 3; ++ky) {
            const int yy = y + ky - 1;
            const bool oky = (unsigned)yy < 256u;
#pragma unroll
            for (int kx = 0; kx < 3; ++kx) {
                const int xv = px + kx - 1;
                float v = 0.0f;   // SAME zero padding
                if (oky && (unsigned)xv < 256u)
                    v = fmaf(zp[(yy << 8) + xv], s1, t1);
                bM[ky * 3 + kx] = (_Float16)v;
            }
        }
    }

    const size_t obase = ((size_t)n << 22) + (size_t)((y << 8) + x0 + (lane & 15));

#pragma unroll
    for (int ct = 0; ct < 4; ++ct) {
        // ---- A matrix (16 x 32, f16): lane<16 -> M=lane, halves 0..15 = K0..15
        // (only K0..8 nonzero); lane>=16 -> M=lane-16, halves 0..15 =
        // K16..31 (zero).
        v16h aM;
#pragma unroll
        for (int i = 0; i < 16; ++i) aM[i] = (_Float16)0.0f;
        const int ch = ct * 16 + (lane & 15);
        const float* wr = w2 + ch * 9;
        if (lane < 16) {
#pragma unroll
            for (int k = 0; k < 9; ++k) aM[k] = (_Float16)wr[k];
        }

        v8f cM;
#pragma unroll
        for (int r = 0; r < 8; ++r) cM[r] = 0.0f;

        v8f dM = __builtin_amdgcn_wmma_f32_16x16x32_f16(
            /*neg_a=*/false, aM, /*neg_b=*/false, bM,
            /*c_mod=*/(short)0, cM, /*reuse_a=*/false, /*reuse_b=*/false);

        // D: VGPR r -> channel ct*16 + r (lanes 0-15) / +8 (lanes 16-31);
        // N = pixel = lane & 15.
#pragma unroll
        for (int r = 0; r < 8; ++r) {
            const int cc = ct * 16 + r + ((lane >> 4) << 3);
            out[obase + ((size_t)cc << 16)] = fmaf(dM[r], sc[cc], sh[cc]);
        }
    }
}

// ---------------------------------------------------------------------------
// Workspace layout (floats):
//   zr[NPIX] | stat[2] gsum[54] (accums, zeroed each call) | s1t1[2] sc[64] sh[64]
// ---------------------------------------------------------------------------
extern "C" void kernel_launch(void* const* d_in, const int* in_sizes, int n_in,
                              void* d_out, int out_size, void* d_ws, size_t ws_size,
                              hipStream_t stream) {
    const float* x    = (const float*)d_in[0];
    const float* w1   = (const float*)d_in[1];
    const float* b1   = (const float*)d_in[2];
    const float* w2   = (const float*)d_in[3];
    // d_in[4] = conv2_b: cancels exactly under bn2, unused.
    const float* mu   = (const float*)d_in[5];
    const float* sg   = (const float*)d_in[6];
    const float* b1g  = (const float*)d_in[7];
    const float* b1b  = (const float*)d_in[8];
    const float* b2g  = (const float*)d_in[9];
    const float* b2b  = (const float*)d_in[10];

    float* ws    = (float*)d_ws;
    float* zr    = ws;
    float* stat  = ws + NPIX;       // 2
    float* gsum  = stat + 2;        // 54
    float* s1t1  = gsum + 54;       // 2
    float* sc    = s1t1 + 2;        // 64
    float* sh    = sc + 64;         // 64

    k_zero<<<1, 256, 0, stream>>>(stat, 56);
    k_conv1_fuzzy<<<1024, 256, 0, stream>>>(x, w1, b1, mu, sg, zr, stat);
    k_bn1_fin<<<1, 32, 0, stream>>>(stat, b1g, b1b, s1t1);
    k_zgram<<<256, 256, 0, stream>>>(zr, s1t1, gsum);
    k_bn2_fin<<<1, 64, 0, stream>>>(gsum, w2, b2g, b2b, sc, sh);
    k_conv2_wmma<<<8192, 256, 0, stream>>>(zr, w2, s1t1, sc, sh, (float*)d_out);
}
